// SignedGAT_41274635715122
// MI455X (gfx1250) — matrix operand order repaired and checked by reading, not compiled, not run on hardware
//
#include <hip/hip_runtime.h>

// ---------------------------------------------------------------------------
// Signed GAT on MI455X (gfx1250). All dense GEMMs via v_wmma_f32_16x16x32_bf16
// (bf16 in, f32 accumulate). Attention is flash-style: per-row masked-softmax
// stats (max, 1/sum) precomputed, then the attention x features GEMM rebuilds
// 16x32 bf16 A-fragments on the fly in LDS (shared across the 8 waves of a
// block) and never materializes the 3072x3072 attention matrices.
// Layer 2 computes only the 64 gathered tweet_id rows.
// ---------------------------------------------------------------------------

typedef __attribute__((ext_vector_type(16))) __bf16 v16bf;
typedef __attribute__((ext_vector_type(8)))  __bf16 v8bf;
typedef __attribute__((ext_vector_type(8)))  float  v8f;
typedef __bf16 bf16;

#define NN    3072       // nodes
#define DF    300        // features per head
#define DP    304        // DF padded to 16
#define KPW   320        // DF padded to 32 (K dim of first GEMM)
#define HH    4          // heads
#define X2W   1216       // 4*DP, layer-2 feature width
#define NT    64         // gathered output rows

__device__ __forceinline__ float fexp(float x) { return __expf(x); }

// ---- pack embedding: f32 [NN,DF] -> xb bf16 [NN,KPW] and xbT bf16 [DP,NN] --
__global__ void k_pack_x(const float* __restrict__ emb,
                         bf16* __restrict__ xb, bf16* __restrict__ xbT) {
  int idx = blockIdx.x * blockDim.x + threadIdx.x;
  if (idx >= NN * KPW) return;
  int i = idx / KPW, c = idx % KPW;
  float v = (c < DF) ? emb[i * DF + c] : 0.f;
  bf16 b = (bf16)v;
  xb[(size_t)i * KPW + c] = b;
  if (c < DP) xbT[(size_t)c * NN + i] = b;
}

// ---- pack weights transposed+padded: f32 [G*blk, C] -> bf16 [CP, G*blkP] ---
__global__ void k_pack_wT(const float* __restrict__ src, bf16* __restrict__ dst,
                          int G, int blk, int blkP, int C, int CP) {
  int KP = G * blkP;
  int idx = blockIdx.x * blockDim.x + threadIdx.x;
  if (idx >= CP * KP) return;
  int c = idx / KP, kp = idx % KP;
  int g = kp / blkP, rr = kp % blkP;
  float v = (c < C && rr < blk) ? src[(size_t)(g * blk + rr) * C + c] : 0.f;
  dst[(size_t)c * KP + kp] = (bf16)v;
}

// ---- bf16 transpose: src [R,C] -> dst [C,R] -------------------------------
__global__ void k_transpose_bf16(const bf16* __restrict__ src,
                                 bf16* __restrict__ dst, int R, int C) {
  int idx = blockIdx.x * blockDim.x + threadIdx.x;
  if (idx >= R * C) return;
  int i = idx / C, c = idx % C;
  dst[(size_t)c * R + i] = src[(size_t)i * C + c];
}

// ---- bf16 WMMA GEMM: C = A[M,K] * Bt[NP,K]^T, epilogue modes --------------
// mode 0: f32 store         (ldo = NP)
// mode 1: ELU -> bf16 store (ldo, colOff)   [layer-1 head output into x2]
// mode 2: sigmoid -> f32 store, col < colLimit only  [final output]
__global__ void k_gemm_bf16(const bf16* __restrict__ A, const bf16* __restrict__ Bt,
                            int M, int K, int NP, int mode,
                            float* __restrict__ outF, bf16* __restrict__ outB,
                            int ldo, int colOff, int colLimit) {
  int lane = threadIdx.x & 31;
  int wave = threadIdx.x >> 5;
  int nCT = NP >> 4;
  int ct = blockIdx.x * 8 + wave;
  if (ct >= nCT) return;
  int mt = blockIdx.y;
  int r16 = lane & 15, half = lane >> 4;
  const bf16* arow = A  + (size_t)(mt * 16 + r16) * K;   // A lane row = M index
  const bf16* brow = Bt + (size_t)(ct * 16 + r16) * K;   // B lane row = N index
  v8f acc = {};
  for (int k0 = 0; k0 < K; k0 += 32) {
    v8bf alo = *(const v8bf*)(arow + k0 + half * 16);      // K = half*16 + 0..7
    v8bf ahi = *(const v8bf*)(arow + k0 + half * 16 + 8);  // K = half*16 + 8..15
    v16bf av, bv;
#pragma unroll
    for (int p = 0; p < 8; ++p) { av[p] = alo[p]; av[p + 8] = ahi[p]; }
    bv = *(const v16bf*)(brow + k0 + half * 16);           // K = half*16 + 0..15
    acc = __builtin_amdgcn_wmma_f32_16x16x32_bf16(false, av, false, bv,
                                                  (short)0, acc, false, false);
  }
  int col = ct * 16 + (lane & 15);
#pragma unroll
  for (int v = 0; v < 8; ++v) {
    int row = mt * 16 + v + 8 * half;
    float val = acc[v];
    if (mode == 0) {
      outF[(size_t)row * ldo + col] = val;
    } else if (mode == 1) {
      float e = val > 0.f ? val : (fexp(val) - 1.f);        // ELU
      outB[(size_t)row * ldo + colOff + col] = (bf16)e;
    } else {
      if (col < colLimit)
        outF[(size_t)row * ldo + col] = 1.f / (1.f + fexp(-val));  // sigmoid
    }
  }
}

// ---- attention score vectors: s1[i] = h[i,:].a[:DF], s2[i] = h[i,:].a[DF:] -
__global__ void k_rowdot(const float* __restrict__ h, const float* __restrict__ a,
                         float* __restrict__ s1, float* __restrict__ s2, int M) {
  int row = blockIdx.x * 8 + (threadIdx.x >> 5);
  if (row >= M) return;
  int lane = threadIdx.x & 31;
  const float* hr = h + (size_t)row * DP;
  float a1 = 0.f, a2 = 0.f;
  for (int c = lane; c < DF; c += 32) {
    float v = hr[c];
    a1 += v * a[c];
    a2 += v * a[DF + c];
  }
  for (int off = 16; off; off >>= 1) {
    a1 += __shfl_down(a1, off, 32);
    a2 += __shfl_down(a2, off, 32);
  }
  if (lane == 0) { s1[row] = a1; s2[row] = a2; }
}

// ---- masked-softmax row stats (max & 1/sum) for +e and -e -----------------
__global__ void k_row_stats(const float* __restrict__ s1, const float* __restrict__ s2,
                            const unsigned char* __restrict__ adj,
                            const int* __restrict__ rowMap,
                            float* __restrict__ mp, float* __restrict__ ilp,
                            float* __restrict__ mn, float* __restrict__ iln) {
  int slot = blockIdx.x;
  int r = rowMap ? rowMap[slot] : slot;
  float s1r = s1[r];
  const unsigned char* arow = adj + (size_t)r * NN;
  float mP = -1e30f, lP = 0.f, mN = -1e30f, lN = 0.f;
  for (int j = threadIdx.x; j < NN; j += blockDim.x) {
    if (arow[j]) {
      float t = s1r + s2[j];
      float e = t > 0.f ? t : 0.3f * t;                     // LeakyReLU(0.3)
      if (e > mP) { lP = lP * fexp(mP - e) + 1.f; mP = e; } else lP += fexp(e - mP);
      float en = -e;
      if (en > mN) { lN = lN * fexp(mN - en) + 1.f; mN = en; } else lN += fexp(en - mN);
    }
  }
  for (int off = 16; off; off >>= 1) {
    float m2 = __shfl_down(mP, off, 32), l2 = __shfl_down(lP, off, 32);
    float mm = fmaxf(mP, m2);
    lP = lP * fexp(mP - mm) + l2 * fexp(m2 - mm); mP = mm;
    m2 = __shfl_down(mN, off, 32); l2 = __shfl_down(lN, off, 32);
    mm = fmaxf(mN, m2);
    lN = lN * fexp(mN - mm) + l2 * fexp(m2 - mm); mN = mm;
  }
  __shared__ float sm[8][4];
  int lane = threadIdx.x & 31, wv = threadIdx.x >> 5;
  if (lane == 0) { sm[wv][0] = mP; sm[wv][1] = lP; sm[wv][2] = mN; sm[wv][3] = lN; }
  __syncthreads();
  if (threadIdx.x == 0) {
    mP = sm[0][0]; lP = sm[0][1]; mN = sm[0][2]; lN = sm[0][3];
    for (int w = 1; w < 8; ++w) {
      float m2 = sm[w][0], l2 = sm[w][1], mm = fmaxf(mP, m2);
      lP = lP * fexp(mP - mm) + l2 * fexp(m2 - mm); mP = mm;
      m2 = sm[w][2]; l2 = sm[w][3]; mm = fmaxf(mN, m2);
      lN = lN * fexp(mN - mm) + l2 * fexp(m2 - mm); mN = mm;
    }
    mp[slot] = mP; ilp[slot] = 1.f / lP;
    mn[slot] = mN; iln[slot] = 1.f / lN;
  }
}

// ---- fused attention GEMM: hp[slot, 0:DW]=att@x, hp[slot, DW:2DW]=negatt@x -
// xT bf16 [DW, NN]; A-fragments (att weights) built on the fly in LDS.
__global__ void k_attn_gemm(const float* __restrict__ s1, const float* __restrict__ s2,
                            const float* __restrict__ mp, const float* __restrict__ ilp,
                            const float* __restrict__ mn, const float* __restrict__ iln,
                            const unsigned char* __restrict__ adj,
                            const int* __restrict__ rowMap,
                            const bf16* __restrict__ xT, int DW,
                            bf16* __restrict__ hp) {
  __shared__ __align__(32) bf16 lAp[512];   // frag-layout: [consumer_lane*16 + p]
  __shared__ __align__(32) bf16 lAn[512];
  int tid = threadIdx.x;
  int lane = tid & 31, wave = tid >> 5;
  int nCT = DW >> 4;
  int ct = blockIdx.x * 8 + wave;
  bool active = ct < nCT;
  int rowTile = blockIdx.y;

  // staging role: this thread fills elements p0,p0+1 of consumer lane sl
  int sl = tid >> 3;
  int sM = sl & 15, sHalf = sl >> 4;
  int sslot = rowTile * 16 + sM;
  int sr = rowMap ? rowMap[sslot] : sslot;
  float s1r  = s1[sr];
  float smp  = mp[sslot], silp = ilp[sslot];
  float smn  = mn[sslot], siln = iln[sslot];
  const unsigned char* arow = adj + (size_t)sr * NN;

  int half = lane >> 4;
  const bf16* brow = xT + (size_t)((active ? ct : 0) * 16 + (lane & 15)) * NN;

  v8f accP = {}, accN = {};
  for (int j0 = 0; j0 < NN; j0 += 32) {
#pragma unroll
    for (int q = 0; q < 2; ++q) {
      int p = (tid & 7) * 2 + q;
      int K = p + sHalf * 16;                     // contiguous K order, matches B
      int j = j0 + K;
      float apv = 0.f, anv = 0.f;
      if (arow[j]) {
        float t = s1r + s2[j];
        float e = t > 0.f ? t : 0.3f * t;
        apv =  fexp(e - smp)  * silp;              // softmax(+e)
        anv = -fexp(-e - smn) * siln;              // -softmax(-e)
      }
      lAp[sl * 16 + p] = (bf16)apv;
      lAn[sl * 16 + p] = (bf16)anv;
    }
    __syncthreads();
    if (active) {
      v16bf ap = *(const v16bf*)(lAp + lane * 16);
      v16bf an = *(const v16bf*)(lAn + lane * 16);
      v16bf bv = *(const v16bf*)(brow + j0 + half * 16);
      accP = __builtin_amdgcn_wmma_f32_16x16x32_bf16(false, ap, false, bv,
                                                     (short)0, accP, false, false);
      accN = __builtin_amdgcn_wmma_f32_16x16x32_bf16(false, an, false, bv,
                                                     (short)0, accN, false, false);
    }
    __syncthreads();
  }
  if (active) {
    int col = ct * 16 + (lane & 15);
#pragma unroll
    for (int v = 0; v < 8; ++v) {
      int srow = rowTile * 16 + v + 8 * half;
      hp[(size_t)srow * (2 * DW) + col]      = (bf16)accP[v];
      hp[(size_t)srow * (2 * DW) + DW + col] = (bf16)accN[v];
    }
  }
}

// ---------------------------------------------------------------------------
extern "C" void kernel_launch(void* const* d_in, const int* in_sizes, int n_in,
                              void* d_out, int out_size, void* d_ws, size_t ws_size,
                              hipStream_t stream) {
  (void)in_sizes; (void)n_in; (void)out_size; (void)ws_size;
  const float* emb = (const float*)d_in[0];
  const float* Whd = (const float*)d_in[1];   // [4,300,300]
  const float* ahd = (const float*)d_in[2];   // [4,600,1]
  const float* wth = (const float*)d_in[3];   // [4,600,300]
  const float* Wo  = (const float*)d_in[4];   // [1200,300]
  const float* ao  = (const float*)d_in[5];   // [600,1]
  const float* wto = (const float*)d_in[6];   // [2400,300]
  const unsigned char* adj = (const unsigned char*)d_in[7];   // bool [3072,3072]
  const int* tweet = (const int*)d_in[8];     // [64]
  float* out = (float*)d_out;                 // [64,300]

  char* ws = (char*)d_ws;
  size_t off = 0;
  auto alloc = [&](size_t bytes) -> char* {
    char* p = ws + off;
    off = (off + bytes + 255) & ~(size_t)255;
    return p;
  };
  bf16*  xb   = (bf16*)alloc((size_t)NN * KPW * 2);     // GEMM-A view of x
  bf16*  xbT  = (bf16*)alloc((size_t)DP * NN * 2);      // attention-B view of x
  bf16*  WbT  = (bf16*)alloc((size_t)DP * KPW * 2);     // per-head W (reused)
  float* h    = (float*)alloc((size_t)NN * DP * 4);     // h = x@W (reused)
  float* s1   = (float*)alloc((size_t)NN * 4);
  float* s2   = (float*)alloc((size_t)NN * 4);
  float* mp   = (float*)alloc((size_t)NN * 4);
  float* ilp  = (float*)alloc((size_t)NN * 4);
  float* mn   = (float*)alloc((size_t)NN * 4);
  float* iln  = (float*)alloc((size_t)NN * 4);
  bf16*  hp   = (bf16*)alloc((size_t)NN * (2 * DP) * 2);   // [pos|neg] (reused)
  bf16*  wtT  = (bf16*)alloc((size_t)DP * (2 * DP) * 2);   // per-head wtrans
  bf16*  x2   = (bf16*)alloc((size_t)NN * X2W * 2);        // layer-2 input
  bf16*  x2T  = (bf16*)alloc((size_t)X2W * NN * 2);
  bf16*  WoT  = (bf16*)alloc((size_t)DP * X2W * 2);
  bf16*  hp2  = (bf16*)alloc((size_t)NT * (2 * X2W) * 2);
  bf16*  wtoT = (bf16*)alloc((size_t)DP * (2 * X2W) * 2);

  const int B = 256;
  auto blocks = [](long long n) { return (int)((n + 255) / 256); };

  // ---- layer 1 ----
  k_pack_x<<<blocks((long long)NN * KPW), B, 0, stream>>>(emb, xb, xbT);

  for (int hh = 0; hh < HH; ++hh) {
    k_pack_wT<<<blocks((long long)DP * KPW), B, 0, stream>>>(
        Whd + (size_t)hh * DF * DF, WbT, 1, DF, KPW, DF, DP);
    k_gemm_bf16<<<dim3(3, NN / 16), B, 0, stream>>>(        // h = x @ W
        xb, WbT, NN, KPW, DP, 0, h, nullptr, DP, 0, 0);
    k_rowdot<<<NN / 8, B, 0, stream>>>(h, ahd + (size_t)hh * 2 * DF, s1, s2, NN);
    k_row_stats<<<NN, B, 0, stream>>>(s1, s2, adj, nullptr, mp, ilp, mn, iln);
    k_attn_gemm<<<dim3(3, NN / 16), B, 0, stream>>>(        // hp = [att@x | -att@x]
        s1, s2, mp, ilp, mn, iln, adj, nullptr, xbT, DP, hp);
    k_pack_wT<<<blocks((long long)DP * 2 * DP), B, 0, stream>>>(
        wth + (size_t)hh * 2 * DF * DF, wtT, 2, DF, DP, DF, DP);
    k_gemm_bf16<<<dim3(3, NN / 16), B, 0, stream>>>(        // x2[:,hh] = ELU(hp@wt)
        hp, wtT, NN, 2 * DP, DP, 1, nullptr, x2, X2W, hh * DP, 0);
  }

  // ---- layer 2 (only 64 gathered rows need attention output) ----
  k_transpose_bf16<<<blocks((long long)NN * X2W), B, 0, stream>>>(x2, x2T, NN, X2W);
  k_pack_wT<<<blocks((long long)DP * X2W), B, 0, stream>>>(Wo, WoT, 4, DF, DP, DF, DP);
  k_gemm_bf16<<<dim3(3, NN / 16), B, 0, stream>>>(          // h2 = x2 @ W_out
      x2, WoT, NN, X2W, DP, 0, h, nullptr, DP, 0, 0);
  k_rowdot<<<NN / 8, B, 0, stream>>>(h, ao, s1, s2, NN);
  k_row_stats<<<NT, B, 0, stream>>>(s1, s2, adj, tweet, mp, ilp, mn, iln);
  k_attn_gemm<<<dim3(10, NT / 16), B, 0, stream>>>(         // hp2 (gathered rows)
      s1, s2, mp, ilp, mn, iln, adj, tweet, x2T, X2W, hp2);
  k_pack_wT<<<blocks((long long)DP * 2 * X2W), B, 0, stream>>>(
      wto, wtoT, 8, DF, DP, DF, DP);
  k_gemm_bf16<<<dim3(3, NT / 16), B, 0, stream>>>(          // out = sigmoid(hp2@wt)
      hp2, wtoT, NT, 2 * X2W, DP, 2, out, nullptr, DF, 0, DF);
}
